// MultiHeadAttention_82360292868240
// MI455X (gfx1250) — matrix operand order repaired and checked by reading, not compiled, not run on hardware
//
#include <hip/hip_runtime.h>

// ---------------------------------------------------------------- constants
#define B_   2
#define L_   2048
#define H_   16
#define D_   1024
#define Q_   64
#define BL_  (B_ * L_)
#define LNEPS 1e-5f
#define SCL  0.125f  // 1/sqrt(64)

typedef __attribute__((ext_vector_type(16))) __bf16 v16bf;
typedef __attribute__((ext_vector_type(8)))  __bf16 v8bf;
typedef __attribute__((ext_vector_type(8)))  float  v8f;

static __device__ __forceinline__ __bf16 f2bf(float f) {
  union { float f; unsigned u; } v; v.f = f;
  unsigned r = (v.u + 0x7FFFu + ((v.u >> 16) & 1u)) >> 16;
  union { unsigned short s; __bf16 b; } o; o.s = (unsigned short)r;
  return o.b;
}

// A-fragment (16x32 bf16): lane row = lane&15.
// lane<16 : elems 0..15 = K k0+0..15
// lane>=16: elems 0..15 = K k0+16..31
static __device__ __forceinline__ v16bf ld_a(const __bf16* rowPtr, int k0, int hi) {
  const v8bf lo = *(const v8bf*)(rowPtr + k0 + hi * 16);
  const v8bf hp = *(const v8bf*)(rowPtr + k0 + hi * 16 + 8);
  v16bf a;
#pragma unroll
  for (int i = 0; i < 8; ++i) { a[i] = lo[i]; a[i + 8] = hp[i]; }
  return a;
}

#define WMMA_BF16(A, Bv, C) \
  __builtin_amdgcn_wmma_f32_16x16x32_bf16(false, (A), false, (Bv), (short)0, (C), false, false)

// C = A(16xK, row-major bf16, pre-offset to tile row 0) x B(Kx64) where Bt is
// stored column-contiguous: Bt[n][k], ldb elements per column. 4 N-subtiles.
static __device__ __forceinline__ void gemm16x64(const __bf16* A, int lda,
                                                 const __bf16* Bt, int ldb,
                                                 int K, int lane, v8f acc[4]) {
  const int hi = (lane >> 4) & 1, nl = lane & 15;
  const __bf16* ap  = A  + (size_t)nl * lda;
  const __bf16* bp0 = Bt + (size_t)(nl +  0) * ldb;
  const __bf16* bp1 = Bt + (size_t)(nl + 16) * ldb;
  const __bf16* bp2 = Bt + (size_t)(nl + 32) * ldb;
  const __bf16* bp3 = Bt + (size_t)(nl + 48) * ldb;
  for (int k = 0; k < K; k += 32) {
    __builtin_prefetch(ap + k + 128, 0, 0);  // global_prefetch_b8 (GL2 warm)
    v16bf a  = ld_a(ap, k, hi);
    v16bf b0 = *(const v16bf*)(bp0 + k + hi * 16);
    v16bf b1 = *(const v16bf*)(bp1 + k + hi * 16);
    v16bf b2 = *(const v16bf*)(bp2 + k + hi * 16);
    v16bf b3 = *(const v16bf*)(bp3 + k + hi * 16);
    acc[0] = WMMA_BF16(a, b0, acc[0]);
    acc[1] = WMMA_BF16(a, b1, acc[1]);
    acc[2] = WMMA_BF16(a, b2, acc[2]);
    acc[3] = WMMA_BF16(a, b3, acc[3]);
  }
}

// ---------------------------------------------------------------- converts
__global__ void cvt_x_kernel(const float* __restrict__ x, __bf16* __restrict__ xb, int n) {
  int i = blockIdx.x * blockDim.x + threadIdx.x;
  if (i < n) xb[i] = f2bf(x[i]);
}

// w [H][D][Q] -> wt [H][Q][D]  (column-contiguous for GEMM B-fragments)
__global__ void cvt_wt_kernel(const float* __restrict__ w, __bf16* __restrict__ wt) {
  int i = blockIdx.x * blockDim.x + threadIdx.x;
  if (i >= H_ * D_ * Q_) return;
  int h = i / (Q_ * D_);
  int r = i % (Q_ * D_);
  int q = r / D_;
  int m = r % D_;
  wt[i] = f2bf(w[((size_t)h * D_ + m) * Q_ + q]);
}

// w_o [H][Q][D] -> wt [D (cols m)][H*Q (k)]
__global__ void cvt_wo_kernel(const float* __restrict__ w, __bf16* __restrict__ wt) {
  int i = blockIdx.x * blockDim.x + threadIdx.x;
  if (i >= D_ * D_) return;
  int m = i / D_;
  int k = i % D_;
  int h = k / Q_;
  int q = k % Q_;
  wt[i] = f2bf(w[((size_t)h * Q_ + q) * D_ + m]);
}

// ---------------------------------------------------------------- QKV GEMM
// one wave -> 16 rows x 64 cols of one head's q/k/v
__global__ void qkv_gemm_kernel(const __bf16* __restrict__ xb,
                                const __bf16* __restrict__ wqt,
                                const __bf16* __restrict__ wkt,
                                const __bf16* __restrict__ wvt,
                                __bf16* __restrict__ qb,
                                __bf16* __restrict__ kb,
                                __bf16* __restrict__ vtb) {
  const int lane  = threadIdx.x & 31;
  const int gwave = (blockIdx.x * blockDim.x + threadIdx.x) >> 5;
  const int tilesPerTensor = H_ * (BL_ / 16);
  if (gwave >= 3 * tilesPerTensor) return;
  const int tensor = gwave / tilesPerTensor;
  const int rem    = gwave % tilesPerTensor;
  const int h      = rem / (BL_ / 16);
  const int mt     = rem % (BL_ / 16);
  const int row0   = mt * 16;

  const __bf16* wt = (tensor == 0) ? wqt : (tensor == 1) ? wkt : wvt;
  v8f acc[4] = {};
  gemm16x64(xb + (size_t)row0 * D_, D_, wt + (size_t)h * Q_ * D_, D_, D_, lane, acc);

  const int b  = row0 / L_;
  const int l0 = row0 % L_;
  const int hi = (lane >> 4) & 1, nl = lane & 15;

  if (tensor < 2) {
    __bf16* dst = ((tensor == 0) ? qb : kb) + (((size_t)b * H_ + h) * L_ + l0) * Q_;
#pragma unroll
    for (int t = 0; t < 4; ++t)
#pragma unroll
      for (int r = 0; r < 8; ++r) {
        int m = r + 8 * hi;
        dst[(size_t)m * Q_ + t * 16 + nl] = f2bf(acc[t][r]);
      }
  } else {
    // V stored transposed: vt[b][h][q][l]
    __bf16* dst = vtb + ((size_t)b * H_ + h) * Q_ * L_;
#pragma unroll
    for (int t = 0; t < 4; ++t)
#pragma unroll
      for (int r = 0; r < 8; ++r) {
        int m = r + 8 * hi;
        int q = t * 16 + nl;
        dst[(size_t)q * L_ + l0 + m] = f2bf(acc[t][r]);
      }
  }
}

// ---------------------------------------------------------------- attention
// one wave -> one (b,h) 16-row i-tile; flash-style streaming over j
__global__ void attn_kernel(const __bf16* __restrict__ qb,
                            const __bf16* __restrict__ kb,
                            const __bf16* __restrict__ vtb,
                            __bf16* __restrict__ att) {
  __shared__ __bf16 Pstage[4][16 * 32];  // per-wave P tile staging
  const int wib  = threadIdx.x >> 5;
  const int lane = threadIdx.x & 31;
  const int gwave = blockIdx.x * (blockDim.x >> 5) + wib;
  if (gwave >= B_ * H_ * (L_ / 16)) return;
  const int it = gwave % (L_ / 16);
  const int bh = gwave / (L_ / 16);
  const int i0 = it * 16;
  const int b  = bh / H_;
  const int h  = bh % H_;

  const __bf16* Qp = qb  + ((size_t)bh * L_ + i0) * Q_;
  const __bf16* Kp = kb  + (size_t)bh * L_ * Q_;
  const __bf16* Vp = vtb + (size_t)bh * Q_ * L_;
  const int hi = (lane >> 4) & 1, nl = lane & 15;

  const __bf16* qrow = Qp + (size_t)nl * Q_;
  const v16bf a0 = ld_a(qrow, 0, hi);
  const v16bf a1 = ld_a(qrow, 32, hi);

  v8f   O[4] = {};
  float mrow[8], lrow[8];
#pragma unroll
  for (int r = 0; r < 8; ++r) { mrow[r] = -3.0e38f; lrow[r] = 0.0f; }
  __bf16* P = &Pstage[wib][0];

  for (int j = 0; j < L_; j += 32) {
    // ---- S = Q K^T for cols [j, j+32)
    const __bf16* k0p = Kp + (size_t)(j      + nl) * Q_ + hi * 16;
    const __bf16* k1p = Kp + (size_t)(j + 16 + nl) * Q_ + hi * 16;
    v16bf bk00 = *(const v16bf*)(k0p);       // K-dim q 0..31
    v16bf bk01 = *(const v16bf*)(k0p + 32);  // K-dim q 32..63
    v16bf bk10 = *(const v16bf*)(k1p);
    v16bf bk11 = *(const v16bf*)(k1p + 32);
    v8f c0 = {};
    c0 = WMMA_BF16(a0, bk00, c0);
    c0 = WMMA_BF16(a1, bk01, c0);
    v8f c1 = {};
    c1 = WMMA_BF16(a0, bk10, c1);
    c1 = WMMA_BF16(a1, bk11, c1);

    // ---- online softmax + stage P (bf16) into LDS in A-fragment layout
#pragma unroll
    for (int r = 0; r < 8; ++r) {
      float s0 = c0[r] * SCL, s1 = c1[r] * SCL;
      float t = fmaxf(s0, s1);
      t = fmaxf(t, __shfl_xor(t, 1, 32));
      t = fmaxf(t, __shfl_xor(t, 2, 32));
      t = fmaxf(t, __shfl_xor(t, 4, 32));
      t = fmaxf(t, __shfl_xor(t, 8, 32));
      float mnew  = fmaxf(mrow[r], t);
      float alpha = __expf(mrow[r] - mnew);
      float p0 = __expf(s0 - mnew), p1 = __expf(s1 - mnew);
      float ls = p0 + p1;
      ls += __shfl_xor(ls, 1, 32);
      ls += __shfl_xor(ls, 2, 32);
      ls += __shfl_xor(ls, 4, 32);
      ls += __shfl_xor(ls, 8, 32);
      lrow[r] = lrow[r] * alpha + ls;
      mrow[r] = mnew;
#pragma unroll
      for (int t4 = 0; t4 < 4; ++t4) O[t4][r] *= alpha;
      int m = r + 8 * hi;
      P[m * 32 + nl]      = f2bf(p0);
      P[m * 32 + 16 + nl] = f2bf(p1);
    }
    asm volatile("s_wait_dscnt 0" ::: "memory");

    // ---- reload P as an A fragment (16x32)
    const __bf16* prow = P + (size_t)nl * 32;
    v16bf pa = ld_a(prow, 0, hi);

    // ---- O += P * V   (B fragments contiguous from transposed V)
#pragma unroll
    for (int t4 = 0; t4 < 4; ++t4) {
      v16bf bv = *(const v16bf*)(Vp + (size_t)(t4 * 16 + nl) * L_ + j + hi * 16);
      O[t4] = WMMA_BF16(pa, bv, O[t4]);
    }
  }

  // ---- finalize: divide by denominator, store att as [B][L][H][Q]
#pragma unroll
  for (int r = 0; r < 8; ++r) {
    float inv = 1.0f / lrow[r];
    int m = r + 8 * hi;
    size_t rowbase = (((size_t)b * L_ + i0 + m) * H_ + h) * Q_;
#pragma unroll
    for (int t4 = 0; t4 < 4; ++t4)
      att[rowbase + t4 * 16 + nl] = f2bf(O[t4][r] * inv);
  }
}

// ---------------------------------------------------------------- out proj
__global__ void out_gemm_kernel(const __bf16* __restrict__ attb,
                                const __bf16* __restrict__ wot,
                                float* __restrict__ outf) {
  const int lane  = threadIdx.x & 31;
  const int gwave = (blockIdx.x * blockDim.x + threadIdx.x) >> 5;
  if (gwave >= (BL_ / 16) * (D_ / 64)) return;
  const int mt = gwave / (D_ / 64);
  const int nt = gwave % (D_ / 64);
  v8f acc[4] = {};
  gemm16x64(attb + (size_t)(mt * 16) * D_, D_, wot + (size_t)(nt * 64) * D_, D_, D_, lane, acc);
  const int hi = (lane >> 4) & 1, nl = lane & 15;
#pragma unroll
  for (int t = 0; t < 4; ++t)
#pragma unroll
    for (int r = 0; r < 8; ++r) {
      int m = r + 8 * hi;
      outf[(size_t)(mt * 16 + m) * D_ + nt * 64 + t * 16 + nl] = acc[t][r];
    }
}

// ---------------------------------------------------------------- LN
__global__ void ln_kernel(const float* __restrict__ x, const float* __restrict__ outf,
                          const float* __restrict__ gamma, const float* __restrict__ beta,
                          float* __restrict__ y) {
  const int row  = blockIdx.x;           // B*L rows
  const int tid  = threadIdx.x;          // 256 threads
  const int lane = tid & 31, wid = tid >> 5;
  __shared__ float sm[8], sq[8], bc[2];

  float v[4], s = 0.0f, ss = 0.0f;
#pragma unroll
  for (int i = 0; i < 4; ++i) {
    int c = tid + i * 256;
    float t = x[(size_t)row * D_ + c] + outf[(size_t)row * D_ + c];
    v[i] = t; s += t; ss += t * t;
  }
#pragma unroll
  for (int msk = 1; msk < 32; msk <<= 1) {
    s  += __shfl_xor(s,  msk, 32);
    ss += __shfl_xor(ss, msk, 32);
  }
  if (lane == 0) { sm[wid] = s; sq[wid] = ss; }
  __syncthreads();
  if (tid == 0) {
    float ts = 0.0f, tq = 0.0f;
    for (int i = 0; i < 8; ++i) { ts += sm[i]; tq += sq[i]; }
    float mu  = ts / (float)D_;
    float var = tq / (float)D_ - mu * mu;
    bc[0] = mu;
    bc[1] = __frsqrt_rn(var + LNEPS);
  }
  __syncthreads();
  const float mu = bc[0], rstd = bc[1];
#pragma unroll
  for (int i = 0; i < 4; ++i) {
    int c = tid + i * 256;
    y[(size_t)row * D_ + c] = (v[i] - mu) * rstd * gamma[c] + beta[c];
  }
}

// ---------------------------------------------------------------- launch
extern "C" void kernel_launch(void* const* d_in, const int* in_sizes, int n_in,
                              void* d_out, int out_size, void* d_ws, size_t ws_size,
                              hipStream_t stream) {
  const float* x     = (const float*)d_in[0];
  const float* w_q   = (const float*)d_in[1];
  const float* w_k   = (const float*)d_in[2];
  const float* w_v   = (const float*)d_in[3];
  const float* w_o   = (const float*)d_in[4];
  const float* gamma = (const float*)d_in[5];
  const float* beta  = (const float*)d_in[6];
  float* y = (float*)d_out;

  char* ws = (char*)d_ws;
  size_t off = 0;
  auto take = [&](size_t bytes) -> char* {
    char* p = ws + off;
    off = (off + bytes + 255) & ~(size_t)255;
    return p;
  };
  __bf16* x_bf  = (__bf16*)take((size_t)BL_ * D_ * 2);
  __bf16* wq_t  = (__bf16*)take((size_t)H_ * Q_ * D_ * 2);
  __bf16* wk_t  = (__bf16*)take((size_t)H_ * Q_ * D_ * 2);
  __bf16* wv_t  = (__bf16*)take((size_t)H_ * Q_ * D_ * 2);
  __bf16* wo_t  = (__bf16*)take((size_t)D_ * D_ * 2);
  __bf16* q_bf  = (__bf16*)take((size_t)B_ * H_ * L_ * Q_ * 2);
  __bf16* k_bf  = (__bf16*)take((size_t)B_ * H_ * L_ * Q_ * 2);
  __bf16* vt_bf = (__bf16*)take((size_t)B_ * H_ * Q_ * L_ * 2);
  __bf16* attbf = (__bf16*)take((size_t)BL_ * D_ * 2);
  float*  out_f = (float*)take((size_t)BL_ * D_ * 4);
  (void)ws_size; (void)in_sizes; (void)n_in; (void)out_size;

  // converts
  cvt_x_kernel<<<(BL_ * D_ + 255) / 256, 256, 0, stream>>>(x, x_bf, BL_ * D_);
  cvt_wt_kernel<<<(H_ * D_ * Q_ + 255) / 256, 256, 0, stream>>>(w_q, wq_t);
  cvt_wt_kernel<<<(H_ * D_ * Q_ + 255) / 256, 256, 0, stream>>>(w_k, wk_t);
  cvt_wt_kernel<<<(H_ * D_ * Q_ + 255) / 256, 256, 0, stream>>>(w_v, wv_t);
  cvt_wo_kernel<<<(D_ * D_ + 255) / 256, 256, 0, stream>>>(w_o, wo_t);

  // QKV: 3 * H * (BL/16) = 12288 waves; 4 waves per 128-thread block
  qkv_gemm_kernel<<<12288 / 4, 128, 0, stream>>>(x_bf, wq_t, wk_t, wv_t, q_bf, k_bf, vt_bf);

  // attention: B*H*(L/16) = 4096 waves
  attn_kernel<<<4096 / 4, 128, 0, stream>>>(q_bf, k_bf, vt_bf, attbf);

  // out projection: (BL/16)*(D/64) = 4096 waves
  out_gemm_kernel<<<4096 / 4, 128, 0, stream>>>(attbf, wo_t, out_f);

  // residual + layernorm
  ln_kernel<<<BL_, 256, 0, stream>>>(x, out_f, gamma, beta, y);
}
